// NMT_12987981103682
// MI455X (gfx1250) — matrix operand order repaired and checked by reading, not compiled, not run on hardware
//
#include <hip/hip_runtime.h>
#include <stdint.h>

// ---------------------------------------------------------------------------
// Types for CDNA5 WMMA (wave32): D(f32 16x16) = A(bf16 16x32) * B(bf16 32x16) + C
// ---------------------------------------------------------------------------
typedef __attribute__((ext_vector_type(16))) __bf16 v16bf;
typedef __attribute__((ext_vector_type(8)))  float  v8f;

union FragBF { uint32_t u[8]; uint4 q[2]; v16bf v; };

__device__ __forceinline__ uint16_t f2bf(float f) {
    uint32_t u = __float_as_uint(f);
    uint32_t r = (u + 0x7FFFu + ((u >> 16) & 1u)) >> 16;
    return (uint16_t)r;
}
__device__ __forceinline__ float bf2f(uint16_t h) {
    return __uint_as_float(((uint32_t)h) << 16);
}
__device__ __forceinline__ float sigmf(float x) { return 1.0f / (1.0f + __expf(-x)); }

// Problem constants
#define E_   512
#define H_   1024
#define L_   4
#define B_   64
#define SS_  48
#define TT_  48
#define G4H  4096          // 4*H
#define BH   (B_*H_)       // 65536
#define NROWS ((TT_-1)*B_) // 3008 readout rows
#define VCHUNK 128
#define NCHUNKS (32000/VCHUNK) // 250

// ---------------------------------------------------------------------------
// Weight pack: f32 [N,Ksrc] row-major -> bf16 WMMA-B fragment order
// layout: [(tn*Ktiles+tk)*32 + lane]*8 u32 ; lane = (n%16) + 16*(kin>=16),
// vgpr v holds K = kbase + 2v, 2v+1
// ---------------------------------------------------------------------------
__global__ void nmt_pack_w(const float* __restrict__ src, int Nrows, int Ksrc,
                           uint16_t* __restrict__ dst, int Ktiles, int koff) {
    size_t idx = (size_t)blockIdx.x * 256 + threadIdx.x;
    if (idx >= (size_t)Nrows * Ksrc) return;
    int n = (int)(idx / Ksrc), k = (int)(idx % Ksrc);
    int kg = k + koff;
    int tn = n >> 4, tk = kg >> 5, kin = kg & 31;
    int lane = (n & 15) + ((kin >> 4) << 4);
    int v  = (kin & 15) >> 1;
    int hi = kin & 1;
    size_t o = ((((size_t)tn * Ktiles + tk) * 32 + lane) * 8 + v) * 2 + hi;
    dst[o] = f2bf(src[idx]);
}

// ---------------------------------------------------------------------------
// A-fragment (this code's contiguous k order, as in the B pack): two contiguous 16B spans per lane:
//   span0: K = 16*khalf .. +7   (VGPR 0..3)
//   span1: K = 16*khalf+8 ..+7  (VGPR 4..7)
// => two global_load_b128 per lane.
// ---------------------------------------------------------------------------
__device__ __forceinline__ void load_a_frag(FragBF& a, const uint16_t* rowp, int kloc) {
    a.q[0] = *(const uint4*)(const void*)(rowp + kloc);
    a.q[1] = *(const uint4*)(const void*)(rowp + kloc + 8);
}
__device__ __forceinline__ void load_b_frag(FragBF& b, const uint32_t* bp) {
    b.q[0] = *(const uint4*)(const void*)(bp);
    b.q[1] = *(const uint4*)(const void*)(bp + 4);
}

// ---------------------------------------------------------------------------
// Fused multi-phase-K WMMA GEMM: C[64,N] = concat(A1,A2,A3)[64,Ktot] @ Bp^T
// A*: bf16 row-major (stride = its own K). Bp: fragment-packed bf16.
// grid.x = N/64, block = 256 (8 waves). wave: mt = w&3, n-pair = w>>2.
// Phase-split K loop: no per-iteration source branching.
// ---------------------------------------------------------------------------
__global__ __launch_bounds__(256)
void nmt_wmma_gemm(const uint16_t* __restrict__ A1, int K1,
                   const uint16_t* __restrict__ A2, int K2,
                   const uint16_t* __restrict__ A3, int K3,
                   const uint32_t* __restrict__ Bp,
                   float* __restrict__ C, int N) {
    const int lane = threadIdx.x & 31;
    const int wv   = threadIdx.x >> 5;
    const int mt   = wv & 3;
    const int nt0  = blockIdx.x * 4 + (wv >> 2) * 2;
    const int Ktiles = (K1 + K2 + K3) >> 5;
    const int mloc  = mt * 16 + (lane & 15);
    const int khalf = lane >> 4;

    v8f acc0 = {0.f,0.f,0.f,0.f,0.f,0.f,0.f,0.f};
    v8f acc1 = acc0;

    const uint16_t* Asrc[3] = { A1, A2, A3 };
    const int       Ksrc[3] = { K1, K2, K3 };

    const uint32_t* b0p = Bp + (((size_t)nt0      * Ktiles) * 32 + lane) * 8;
    const uint32_t* b1p = Bp + (((size_t)(nt0+1)  * Ktiles) * 32 + lane) * 8;

    int ktg = 0;
    for (int ph = 0; ph < 3; ++ph) {
        const int Kloc = Ksrc[ph];
        if (Kloc == 0) continue;
        const int nkt = Kloc >> 5;
        const uint16_t* rowp = Asrc[ph] + (size_t)mloc * Kloc + 16 * khalf;
        for (int kk = 0; kk < nkt; ++kk, ++ktg) {
            FragBF a, b0, b1;
            load_a_frag(a, rowp, kk << 5);
            load_b_frag(b0, b0p + (size_t)ktg * 256);
            load_b_frag(b1, b1p + (size_t)ktg * 256);
            if (ktg + 1 < Ktiles) __builtin_prefetch(b0p + (size_t)(ktg + 1) * 256, 0, 1);

            acc0 = __builtin_amdgcn_wmma_f32_16x16x32_bf16(false, a.v, false, b0.v,
                                                           (short)0, acc0, false, false);
            acc1 = __builtin_amdgcn_wmma_f32_16x16x32_bf16(false, a.v, false, b1.v,
                                                           (short)0, acc1, false, false);
        }
    }
    const int col = nt0 * 16 + (lane & 15);
#pragma unroll
    for (int r = 0; r < 8; ++r) {
        int row = mt * 16 + r + 8 * khalf;   // C layout: VGPR r -> M = r (+8 hi lanes)
        C[(size_t)row * N + col]      = acc0[r];
        C[(size_t)row * N + col + 16] = acc1[r];
    }
}

// ---------------------------------------------------------------------------
// LSTM pointwise: gates[64,4096] (+bih+bhh) -> c,h ; h also stored bf16
// ---------------------------------------------------------------------------
__global__ void nmt_lstm_pw(const float* __restrict__ gates,
                            const float* __restrict__ bih, const float* __restrict__ bhh,
                            float* __restrict__ c, float* __restrict__ h_f32,
                            uint16_t* __restrict__ h_bf, float* __restrict__ h_copy) {
    int idx = blockIdx.x * 256 + threadIdx.x;     // < 65536
    int b = idx >> 10, n = idx & 1023;
    const float* g = gates + (size_t)b * G4H;
    float gi = g[n]      + bih[n]      + bhh[n];
    float gf = g[H_+n]   + bih[H_+n]   + bhh[H_+n];
    float gg = g[2*H_+n] + bih[2*H_+n] + bhh[2*H_+n];
    float go = g[3*H_+n] + bih[3*H_+n] + bhh[3*H_+n];
    float cn = sigmf(gf) * c[idx] + sigmf(gi) * tanhf(gg);
    float hn = sigmf(go) * tanhf(cn);
    c[idx] = cn; h_f32[idx] = hn; h_bf[idx] = f2bf(hn);
    if (h_copy) h_copy[idx] = hn;
}

// ---------------------------------------------------------------------------
// Attention (per batch b): scores->softmax->ctx (bf16)
// ---------------------------------------------------------------------------
__global__ __launch_bounds__(256)
void nmt_attention(const float* __restrict__ enc_out,  // [48][64][1024]
                   const float* __restrict__ h_top,    // [64][1024]
                   uint16_t* __restrict__ ctx_bf) {    // [64][1024]
    __shared__ float sc[SS_];
    const int b = blockIdx.x;
    const int lane = threadIdx.x & 31;
    const int wv = threadIdx.x >> 5;
    for (int s = wv; s < SS_; s += 8) {
        float acc = 0.f;
        const float* e = enc_out + ((size_t)s * B_ + b) * H_;
        const float* h = h_top + (size_t)b * H_;
        for (int k = lane; k < H_; k += 32) acc += e[k] * h[k];
#pragma unroll
        for (int off = 16; off > 0; off >>= 1) acc += __shfl_xor(acc, off, 32);
        if (lane == 0) sc[s] = acc;
    }
    __syncthreads();
    if (threadIdx.x == 0) {
        float m = sc[0];
        for (int s = 1; s < SS_; ++s) m = fmaxf(m, sc[s]);
        float sum = 0.f;
        for (int s = 0; s < SS_; ++s) { sc[s] = __expf(sc[s] - m); sum += sc[s]; }
        float inv = 1.f / sum;
        for (int s = 0; s < SS_; ++s) sc[s] *= inv;
    }
    __syncthreads();
    for (int k = threadIdx.x; k < H_; k += 256) {
        float cx = 0.f;
        for (int s = 0; s < SS_; ++s) cx += sc[s] * enc_out[((size_t)s * B_ + b) * H_ + k];
        ctx_bf[(size_t)b * H_ + k] = f2bf(cx);
    }
}

// tanh(att_pre) -> bf16 att slot
__global__ void nmt_tanh_bf(const float* __restrict__ src, uint16_t* __restrict__ dst) {
    int i = blockIdx.x * 256 + threadIdx.x;
    dst[i] = f2bf(tanhf(src[i]));
}

// ---------------------------------------------------------------------------
// Readout: WMMA GEMM chunk [16 rows x 128 vocab] with streamed (max, sumexp)
// grid = (250 vocab chunks, 188 row tiles), 8 waves each own one 16-col tile.
// ---------------------------------------------------------------------------
__global__ __launch_bounds__(256)
void nmt_readout_lse(const uint16_t* __restrict__ A,   // [3008][1024] bf16
                     const uint32_t* __restrict__ Bp,  // packed readout W
                     float* __restrict__ partials) {   // [3008][250][2]
    __shared__ float tile[16 * VCHUNK];
    const int lane = threadIdx.x & 31;
    const int wv = threadIdx.x >> 5;
    const int chunk = blockIdx.x;
    const int rt = blockIdx.y;
    const int Ktiles = H_ / 32;                 // 32
    const int ntg = chunk * 8 + wv;             // global 16-col tile (0..1999)
    const int khalf = lane >> 4;
    const uint16_t* rowp = A + (size_t)(rt * 16 + (lane & 15)) * H_ + 16 * khalf;
    const uint32_t* bpb = Bp + (((size_t)ntg * Ktiles) * 32 + lane) * 8;

    v8f acc = {0.f,0.f,0.f,0.f,0.f,0.f,0.f,0.f};
    for (int kt = 0; kt < Ktiles; ++kt) {
        FragBF a, b;
        load_a_frag(a, rowp, kt << 5);
        load_b_frag(b, bpb + (size_t)kt * 256);
        if (kt + 1 < Ktiles) __builtin_prefetch(bpb + (size_t)(kt + 1) * 256, 0, 1);
        acc = __builtin_amdgcn_wmma_f32_16x16x32_bf16(false, a.v, false, b.v,
                                                      (short)0, acc, false, false);
    }
#pragma unroll
    for (int r = 0; r < 8; ++r)
        tile[(r + 8 * khalf) * VCHUNK + wv * 16 + (lane & 15)] = acc[r];
    __syncthreads();
    if (threadIdx.x < 16) {
        int row = threadIdx.x;
        const float* rp = &tile[row * VCHUNK];
        float m = -3.0e38f;
        for (int cJ = 0; cJ < VCHUNK; ++cJ) m = fmaxf(m, rp[cJ]);
        float s = 0.f;
        for (int cJ = 0; cJ < VCHUNK; ++cJ) s += __expf(rp[cJ] - m);
        size_t rowg = (size_t)rt * 16 + row;
        partials[(rowg * NCHUNKS + chunk) * 2 + 0] = m;
        partials[(rowg * NCHUNKS + chunk) * 2 + 1] = s;
    }
}

__global__ void nmt_lse_reduce(const float* __restrict__ partials, float* __restrict__ lse) {
    int row = blockIdx.x * 256 + threadIdx.x;
    if (row >= NROWS) return;
    const float* p = partials + (size_t)row * NCHUNKS * 2;
    float M = -3.0e38f, S = 0.f;
    for (int c = 0; c < NCHUNKS; ++c) {
        float m = p[c*2], s = p[c*2+1];
        if (m > M) { S = S * __expf(M - m) + s; M = m; }
        else       { S += s * __expf(m - M); }
    }
    lse[row] = M + __logf(S);
}

// gold logit (bf16 row . f32 readout row) - lse ; masked; atomicAdd into out[b]
__global__ __launch_bounds__(256)
void nmt_gold(const int* __restrict__ tgt_tokens, const uint16_t* __restrict__ att_rows,
              const float* __restrict__ readout_W, const float* __restrict__ lse,
              float* __restrict__ out) {
    int row = blockIdx.x * 8 + (threadIdx.x >> 5);
    int lane = threadIdx.x & 31;
    if (row >= NROWS) return;
    int t = row >> 6, b = row & 63;
    int gold = tgt_tokens[(t + 1) * B_ + b];
    const uint16_t* a = att_rows + (size_t)row * H_;
    const float* w = readout_W + (size_t)gold * H_;
    float acc = 0.f;
    for (int k = lane; k < H_; k += 32) acc += bf2f(a[k]) * w[k];
#pragma unroll
    for (int off = 16; off > 0; off >>= 1) acc += __shfl_xor(acc, off, 32);
    if (lane == 0 && gold != 0) atomicAdd(&out[b], acc - lse[row]);
}

// ---------------------------------------------------------------------------
// Small helpers: embedding gather (f32->bf16), zero fills, state broadcast
// ---------------------------------------------------------------------------
__global__ void nmt_embed(const int* __restrict__ toks, const float* __restrict__ emb,
                          uint16_t* __restrict__ dst, int nrows) {
    size_t idx = (size_t)blockIdx.x * 256 + threadIdx.x;
    if (idx >= (size_t)nrows * E_) return;
    int e = (int)(idx & (E_ - 1));
    int r = (int)(idx >> 9);
    int tok = toks[r];
    dst[idx] = f2bf(emb[(size_t)tok * E_ + e]);
}
__global__ void nmt_zero_f32(float* p, size_t n) {
    size_t i = (size_t)blockIdx.x * 256 + threadIdx.x; if (i < n) p[i] = 0.f;
}
__global__ void nmt_zero_u16(uint16_t* p, size_t n) {
    size_t i = (size_t)blockIdx.x * 256 + threadIdx.x; if (i < n) p[i] = 0;
}
__global__ void nmt_bcast_state(const float* __restrict__ h_src, const float* __restrict__ c_src,
                                float* __restrict__ h_dst, float* __restrict__ c_dst,
                                uint16_t* __restrict__ hbf_dst) {
    int idx = blockIdx.x * 256 + threadIdx.x;     // < 4*65536
    int j = idx & (BH - 1);
    float hv = h_src[j];
    h_dst[idx] = hv; c_dst[idx] = c_src[j]; hbf_dst[idx] = f2bf(hv);
}

// ---------------------------------------------------------------------------
// Host orchestration
// ---------------------------------------------------------------------------
extern "C" void kernel_launch(void* const* d_in, const int* in_sizes, int n_in,
                              void* d_out, int out_size, void* d_ws, size_t ws_size,
                              hipStream_t stream) {
    (void)in_sizes; (void)n_in; (void)out_size; (void)ws_size;
    const int*   src_tokens = (const int*)  d_in[0];
    const int*   tgt_tokens = (const int*)  d_in[1];
    const float* src_emb    = (const float*)d_in[2];
    const float* tgt_emb    = (const float*)d_in[3];
    const float* enc_Wih0   = (const float*)d_in[4];
    const float* enc_WihR   = (const float*)d_in[5];
    const float* enc_Whh    = (const float*)d_in[6];
    const float* enc_bih    = (const float*)d_in[7];
    const float* enc_bhh    = (const float*)d_in[8];
    const float* dec_Wih0   = (const float*)d_in[9];
    const float* dec_WihR   = (const float*)d_in[10];
    const float* dec_Whh    = (const float*)d_in[11];
    const float* dec_bih    = (const float*)d_in[12];
    const float* dec_bhh    = (const float*)d_in[13];
    const float* att_vec_W  = (const float*)d_in[14];
    const float* readout_W  = (const float*)d_in[15];
    float* out = (float*)d_out;

    // workspace carving (256B aligned)
    char* base = (char*)d_ws; size_t off = 0;
    auto alloc = [&](size_t bytes) -> void* {
        void* p = base + off; off = (off + bytes + 255) & ~(size_t)255; return p;
    };
    uint16_t* pW_enc0   = (uint16_t*)alloc((size_t)G4H * 1536 * 2);
    uint16_t* pW_encR[3]; for (int l=0;l<3;++l) pW_encR[l] = (uint16_t*)alloc((size_t)G4H*2048*2);
    uint16_t* pW_dec0   = (uint16_t*)alloc((size_t)G4H * 2560 * 2);
    uint16_t* pW_decR[3]; for (int l=0;l<3;++l) pW_decR[l] = (uint16_t*)alloc((size_t)G4H*2048*2);
    uint16_t* pW_att    = (uint16_t*)alloc((size_t)H_ * 2048 * 2);
    uint16_t* pW_read   = (uint16_t*)alloc((size_t)32000 * H_ * 2);
    uint16_t* src_x     = (uint16_t*)alloc((size_t)SS_ * B_ * E_ * 2);
    uint16_t* tgt_x     = (uint16_t*)alloc((size_t)(TT_-1) * B_ * E_ * 2);
    float*    enc_h     = (float*)   alloc((size_t)L_ * BH * 4);
    float*    enc_c     = (float*)   alloc((size_t)L_ * BH * 4);
    uint16_t* enc_hbf   = (uint16_t*)alloc((size_t)L_ * BH * 2);
    float*    dec_h     = (float*)   alloc((size_t)L_ * BH * 4);
    float*    dec_c     = (float*)   alloc((size_t)L_ * BH * 4);
    uint16_t* dec_hbf   = (uint16_t*)alloc((size_t)L_ * BH * 2);
    float*    gates     = (float*)   alloc((size_t)B_ * G4H * 4);
    float*    enc_out   = (float*)   alloc((size_t)SS_ * BH * 4);
    uint16_t* ctx_bf    = (uint16_t*)alloc((size_t)BH * 2);
    float*    att_pre   = (float*)   alloc((size_t)BH * 4);
    uint16_t* att_ves   = (uint16_t*)alloc((size_t)TT_ * BH * 2);   // slot 0 = att0 (zeros)
    float*    partials  = (float*)   alloc((size_t)NROWS * NCHUNKS * 2 * 4);
    float*    lse       = (float*)   alloc((size_t)NROWS * 4);

    auto packGrid = [](size_t elems) { return (unsigned)((elems + 255) / 256); };

    // ---- pack weights to WMMA-B fragment bf16 ----
    nmt_pack_w<<<packGrid((size_t)G4H*512), 256, 0, stream>>>(enc_Wih0, G4H, 512, pW_enc0, 1536/32, 0);
    nmt_pack_w<<<packGrid((size_t)G4H*1024),256, 0, stream>>>(enc_Whh,  G4H, 1024, pW_enc0, 1536/32, 512);
    for (int l = 1; l < L_; ++l) {
        nmt_pack_w<<<packGrid((size_t)G4H*1024),256,0,stream>>>(enc_WihR + (size_t)(l-1)*G4H*H_, G4H, 1024, pW_encR[l-1], 2048/32, 0);
        nmt_pack_w<<<packGrid((size_t)G4H*1024),256,0,stream>>>(enc_Whh  + (size_t)l*G4H*H_,     G4H, 1024, pW_encR[l-1], 2048/32, 1024);
    }
    nmt_pack_w<<<packGrid((size_t)G4H*1536),256,0,stream>>>(dec_Wih0, G4H, 1536, pW_dec0, 2560/32, 0);
    nmt_pack_w<<<packGrid((size_t)G4H*1024),256,0,stream>>>(dec_Whh,  G4H, 1024, pW_dec0, 2560/32, 1536);
    for (int l = 1; l < L_; ++l) {
        nmt_pack_w<<<packGrid((size_t)G4H*1024),256,0,stream>>>(dec_WihR + (size_t)(l-1)*G4H*H_, G4H, 1024, pW_decR[l-1], 2048/32, 0);
        nmt_pack_w<<<packGrid((size_t)G4H*1024),256,0,stream>>>(dec_Whh  + (size_t)l*G4H*H_,     G4H, 1024, pW_decR[l-1], 2048/32, 1024);
    }
    nmt_pack_w<<<packGrid((size_t)H_*2048),   256,0,stream>>>(att_vec_W, H_,    2048, pW_att,  2048/32, 0);
    nmt_pack_w<<<packGrid((size_t)32000*H_),  256,0,stream>>>(readout_W, 32000, H_,   pW_read, H_/32,   0);

    // ---- embeddings + state init ----
    nmt_embed<<<packGrid((size_t)SS_*B_*E_),     256,0,stream>>>(src_tokens, src_emb, src_x, SS_*B_);
    nmt_embed<<<packGrid((size_t)(TT_-1)*B_*E_), 256,0,stream>>>(tgt_tokens, tgt_emb, tgt_x, (TT_-1)*B_);
    nmt_zero_f32<<<packGrid((size_t)L_*BH),256,0,stream>>>(enc_h, (size_t)L_*BH);
    nmt_zero_f32<<<packGrid((size_t)L_*BH),256,0,stream>>>(enc_c, (size_t)L_*BH);
    nmt_zero_u16<<<packGrid((size_t)L_*BH),256,0,stream>>>(enc_hbf, (size_t)L_*BH);
    nmt_zero_u16<<<packGrid((size_t)BH),   256,0,stream>>>(att_ves, (size_t)BH);  // att0 = 0
    nmt_zero_f32<<<1, 64, 0, stream>>>(out, (size_t)B_);

    // ---- encoder: 48 steps x 4 layers ----
    for (int t = 0; t < SS_; ++t) {
        const uint16_t* inp = src_x + (size_t)t * B_ * E_;
        int Kin = E_;
        for (int l = 0; l < L_; ++l) {
            const uint32_t* W = (const uint32_t*)((l == 0) ? pW_enc0 : pW_encR[l-1]);
            nmt_wmma_gemm<<<G4H/64, 256, 0, stream>>>(
                inp, Kin, enc_hbf + (size_t)l*BH, H_, nullptr, 0, W, gates, G4H);
            nmt_lstm_pw<<<BH/256, 256, 0, stream>>>(
                gates, enc_bih + (size_t)l*G4H, enc_bhh + (size_t)l*G4H,
                enc_c + (size_t)l*BH, enc_h + (size_t)l*BH, enc_hbf + (size_t)l*BH,
                (l == L_-1) ? (enc_out + (size_t)t*BH) : nullptr);
            inp = enc_hbf + (size_t)l*BH; Kin = H_;
        }
    }

    // ---- broadcast top-layer state to all decoder layers ----
    nmt_bcast_state<<<(L_*BH)/256, 256, 0, stream>>>(
        enc_h + (size_t)3*BH, enc_c + (size_t)3*BH, dec_h, dec_c, dec_hbf);

    // ---- decoder: 47 steps ----
    for (int t = 0; t < TT_-1; ++t) {
        for (int l = 0; l < L_; ++l) {
            if (l == 0) {
                nmt_wmma_gemm<<<G4H/64, 256, 0, stream>>>(
                    tgt_x + (size_t)t * B_ * E_, E_,
                    att_ves + (size_t)t * BH, H_,          // previous att (slot t)
                    dec_hbf, H_,
                    (const uint32_t*)pW_dec0, gates, G4H);
            } else {
                nmt_wmma_gemm<<<G4H/64, 256, 0, stream>>>(
                    dec_hbf + (size_t)(l-1)*BH, H_,
                    dec_hbf + (size_t)l*BH, H_, nullptr, 0,
                    (const uint32_t*)pW_decR[l-1], gates, G4H);
            }
            nmt_lstm_pw<<<BH/256, 256, 0, stream>>>(
                gates, dec_bih + (size_t)l*G4H, dec_bhh + (size_t)l*G4H,
                dec_c + (size_t)l*BH, dec_h + (size_t)l*BH, dec_hbf + (size_t)l*BH, nullptr);
        }
        nmt_attention<<<B_, 256, 0, stream>>>(enc_out, dec_h + (size_t)3*BH, ctx_bf);
        nmt_wmma_gemm<<<H_/64, 256, 0, stream>>>(
            dec_hbf + (size_t)3*BH, H_, ctx_bf, H_, nullptr, 0,
            (const uint32_t*)pW_att, att_pre, H_);
        nmt_tanh_bf<<<BH/256, 256, 0, stream>>>(att_pre, att_ves + (size_t)(t+1)*BH);
    }

    // ---- readout log-softmax (streamed) + gold gather ----
    const uint16_t* att_rows = att_ves + (size_t)BH;   // slots 1..47 => 3008 rows
    nmt_readout_lse<<<dim3(NCHUNKS, NROWS/16), 256, 0, stream>>>(
        att_rows, (const uint32_t*)pW_read, partials);
    nmt_lse_reduce<<<(NROWS + 255)/256, 256, 0, stream>>>(partials, lse);
    nmt_gold<<<(NROWS + 7)/8, 256, 0, stream>>>(tgt_tokens, att_rows, readout_W, lse, out);
}
